// HexRUNet_C_23484881174830
// MI455X (gfx1250) — compile-verified, matrix operand order checked by reading, not run on hardware
//
#include <hip/hip_runtime.h>

typedef __bf16 bf16;
typedef __attribute__((ext_vector_type(16))) __bf16 v16bf;
typedef __attribute__((ext_vector_type(8)))  __bf16 v8bf;
typedef __attribute__((ext_vector_type(8)))  float  v8f;

#define NIMG 1280      // 5 charts * 256 batch
#define NB   256
#define NSLICE 64

// 7 valid hex taps: HEX_MASK zeroes (ky,kx) = (0,0) and (2,2)
__constant__ int c_hky[7] = {0,0,1,1,1,2,2};
__constant__ int c_hkx[7] = {1,2,0,1,2,0,1};

// ---------------------------------------------------------------------------
// Weight packing into WMMA A-fragment layout (16x32 bf16 per fragment):
//   lane 0-15 : M = lane,    elems 0..7 -> K kb+0..7,  elems 8..15 -> K kb+16..23
//   lane 16-31: M = lane-16, elems 0..7 -> K kb+8..15, elems 8..15 -> K kb+24..31
// ---------------------------------------------------------------------------
__global__ void pack_gemm_w(const float* __restrict__ W, bf16* __restrict__ out,
                            int O, int C) {
  int nKt = (C + 31) >> 5;
  int total = (O >> 4) * nKt * 512;
  int i = blockIdx.x * blockDim.x + threadIdx.x;
  if (i >= total) return;
  int e = i & 15, lane = (i >> 4) & 31, f = i >> 9;
  int kt = f % nKt, mt = f / nKt;
  int hi = lane >> 4;
  int m = mt * 16 + (lane & 15);
  int k = kt * 32 + (hi ? 8 : 0) + ((e < 8) ? e : (8 + e));
  float v = (k < C) ? W[m * C + k] : 0.0f;
  out[i] = (bf16)v;
}

__global__ void pack_hex_w(const float* __restrict__ W /*[O,C,3,3]*/,
                           bf16* __restrict__ out, int O, int C) {
  int nKt = C >> 5;
  int nMt = O >> 4;
  int total = 7 * nMt * nKt * 512;
  int i = blockIdx.x * blockDim.x + threadIdx.x;
  if (i >= total) return;
  int e = i & 15, lane = (i >> 4) & 31, f = i >> 9;
  int kt = f % nKt, mt = (f / nKt) % nMt, tap = f / (nKt * nMt);
  int hi = lane >> 4;
  int m = mt * 16 + (lane & 15);
  int k = kt * 32 + (hi ? 8 : 0) + ((e < 8) ? e : (8 + e));
  int ky = c_hky[tap], kx = c_hkx[tap];
  out[i] = (bf16)W[((m * C + k) * 3 + ky) * 3 + kx];
}

// ---------------------------------------------------------------------------
// Stem: hexconv(x, w1) + ReLU -> t0 bf16 NHWC [1280,16,32,16]
// ---------------------------------------------------------------------------
__global__ void stem_kernel(const float* __restrict__ x, const float* __restrict__ w,
                            bf16* __restrict__ t0) {
  int i = blockIdx.x * blockDim.x + threadIdx.x;
  int total = NIMG * 16 * 32 * 16;
  if (i >= total) return;
  int o = i & 15;
  int t = i >> 4;
  int xx = t & 31; t >>= 5;
  int yy = t & 15; t >>= 4;
  int img = t;
  int f = img / NB, b = img % NB;
  float acc = 0.f;
  for (int tp = 0; tp < 7; ++tp) {
    int ky = c_hky[tp], kx = c_hkx[tp];
    int iy = yy + ky - 1;
    if (iy < 0 || iy >= 16) continue;   // pole: zero pad
    int ix = xx + kx - 1;
    int f2 = f, jx = ix;
    if (ix < 0)       { f2 = (f + 4) % 5; jx = 31; }   // prev chart rightmost
    else if (ix >= 32){ f2 = (f + 1) % 5; jx = 0;  }   // next chart leftmost
    const float* xb = x + ((((size_t)(f2 * NB + b) * 3) * 16 + iy) * 32 + jx);
    const float* wb = w + (o * 27) + ky * 3 + kx;
    #pragma unroll
    for (int c = 0; c < 3; ++c) acc += wb[c * 9] * xb[(size_t)c * 512];
  }
  t0[i] = (bf16)fmaxf(acc, 0.f);
}

// ---------------------------------------------------------------------------
// Deterministic BN stats
// ---------------------------------------------------------------------------
__global__ void stats_partial(const bf16* __restrict__ t, float* __restrict__ part,
                              int Npix, int C) {
  int c = blockIdx.x % C;
  int s = blockIdx.x / C;
  int per = (Npix + NSLICE - 1) / NSLICE;
  int p0 = s * per;
  int p1 = p0 + per; if (p1 > Npix) p1 = Npix;
  float sum = 0.f, sq = 0.f;
  for (int p = p0 + (int)threadIdx.x; p < p1; p += blockDim.x) {
    float v = (float)t[(size_t)p * C + c];
    sum += v; sq += v * v;
  }
  __shared__ float s1[256], s2[256];
  int tid = threadIdx.x;
  s1[tid] = sum; s2[tid] = sq;
  __syncthreads();
  for (int off = 128; off > 0; off >>= 1) {
    if (tid < off) { s1[tid] += s1[tid + off]; s2[tid] += s2[tid + off]; }
    __syncthreads();
  }
  if (tid == 0) {
    part[(s * C + c) * 2]     = s1[0];
    part[(s * C + c) * 2 + 1] = s2[0];
  }
}

__global__ void stats_final(const float* __restrict__ part, const float* __restrict__ g,
                            const float* __restrict__ bt, float* __restrict__ mr,
                            int C, float inv_n) {
  int c = blockIdx.x * blockDim.x + threadIdx.x;
  if (c >= C) return;
  float sum = 0.f, sq = 0.f;
  for (int s = 0; s < NSLICE; ++s) {
    sum += part[(s * C + c) * 2];
    sq  += part[(s * C + c) * 2 + 1];
  }
  float mean = sum * inv_n;
  float var  = sq * inv_n - mean * mean;
  float rstd = rsqrtf(var + 1e-5f);
  float sc = g[c] * rstd;
  mr[c * 2]     = sc;
  mr[c * 2 + 1] = bt[c] - mean * sc;
}

// ---------------------------------------------------------------------------
// Elementwise normalize variants (C compile-time -> %C becomes AND)
// ---------------------------------------------------------------------------
template<int C>
__global__ void normalize_plain(const bf16* __restrict__ in, const float* __restrict__ mr,
                                bf16* __restrict__ out, int n, int do_relu) {
  int i = blockIdx.x * blockDim.x + threadIdx.x;
  if (i >= n) return;
  int c = i & (C - 1);
  float v = (float)in[i] * mr[c * 2] + mr[c * 2 + 1];
  if (do_relu) v = fmaxf(v, 0.f);
  out[i] = (bf16)v;
}

template<int C, int H, int W>
__global__ void normalize_pad(const bf16* __restrict__ in, const float* __restrict__ mr,
                              bf16* __restrict__ outP) {
  int i = blockIdx.x * blockDim.x + threadIdx.x;
  constexpr int total = NIMG * H * W * C;
  if (i >= total) return;
  int c = i & (C - 1); int t = i / C;
  int x = t % W; t /= W;
  int y = t % H; int img = t / H;
  float v = (float)in[i] * mr[c * 2] + mr[c * 2 + 1];
  v = fmaxf(v, 0.f);
  outP[(((size_t)img * (H + 2) + (y + 1)) * (W + 2) + (x + 1)) * C + c] = (bf16)v;
}

template<int C, int H, int W>
__global__ void halo_fill(bf16* __restrict__ P) {
  int i = blockIdx.x * blockDim.x + threadIdx.x;
  constexpr int total = NIMG * H * C * 2;
  if (i >= total) return;
  int c = i & (C - 1); int t = i / C;
  int y = t % H; t /= H;
  int side = t & 1; int img = t >> 1;
  int f = img / NB, b = img % NB;
  constexpr int Wp = W + 2, Hp = H + 2;
  if (side == 0) {
    int fs = (f + 4) % 5;
    P[(((size_t)img * Hp + y + 1) * Wp + 0) * C + c] =
        P[(((size_t)(fs * NB + b) * Hp + y + 1) * Wp + W) * C + c];
  } else {
    int fs = (f + 1) % 5;
    P[(((size_t)img * Hp + y + 1) * Wp + (Wp - 1)) * C + c] =
        P[(((size_t)(fs * NB + b) * Hp + y + 1) * Wp + 1) * C + c];
  }
}

__global__ void clear_bf16(bf16* __restrict__ p, int n) {
  int i = blockIdx.x * blockDim.x + threadIdx.x;
  if (i < n) p[i] = (bf16)0.0f;
}

// ---------------------------------------------------------------------------
// WMMA fragment helpers. Chunk validity depends only on (kbase, C): with C a
// compile-time constant and unrolled K-loops this folds to branchless code.
// ---------------------------------------------------------------------------
union V16U { v16bf v; v8bf h[2]; unsigned long long q[4]; };

template<int C>
__device__ __forceinline__ v16bf load_bfrag(const bf16* __restrict__ pix,
                                            int kbase, int hi) {
  V16U u;
  int k0 = kbase + (hi ? 8 : 0);
  if (kbase < C)      u.h[0] = *(const v8bf*)(pix + k0);
  else                { u.q[0] = 0ULL; u.q[1] = 0ULL; }
  if (kbase + 16 < C) u.h[1] = *(const v8bf*)(pix + k0 + 16);
  else                { u.q[2] = 0ULL; u.q[3] = 0ULL; }
  return u.v;
}

__device__ __forceinline__ v8f wmma_bf16(v16bf a, v16bf b, v8f c) {
  return __builtin_amdgcn_wmma_f32_16x16x32_bf16(false, a, false, b, (short)0, c,
                                                 false, false);
}

// ---------------------------------------------------------------------------
// conv1x1 + 2x2 maxpool, fused: one A-fragment feeds 4 independent quadrant
// WMMA chains; elementwise max -> pooled pre-BN bf16.
// act [NIMG,Hin,Win,C] -> out [NIMG,Hin/2,Win/2,O]
// ---------------------------------------------------------------------------
template<int C, int Hin, int Win, int O>
__global__ void __launch_bounds__(256) gemm_conv_pool(
    const bf16* __restrict__ act, const bf16* __restrict__ wp, bf16* __restrict__ out) {
  constexpr int nKt = (C + 31) / 32;
  constexpr int nMt = O / 16;
  constexpr int Ho = Hin / 2, Wo = Win / 2;
  constexpr int Npool = NIMG * Ho * Wo;
  constexpr int nNt = Npool / 16;
  int lane = threadIdx.x & 31;
  int hi = lane >> 4;
  int wv = (blockIdx.x * blockDim.x + threadIdx.x) >> 5;
  if (wv >= nMt * nNt) return;
  int mt = wv % nMt;
  int nt = wv / nMt;
  int gp = nt * 16 + (lane & 15);
  int img = gp / (Ho * Wo);
  int r = gp - img * (Ho * Wo);
  int py = r / Wo, px = r - (r / Wo) * Wo;
  const bf16* base0 = act + (((size_t)img * Hin + 2 * py) * Win + 2 * px) * C;
  const bf16* b1p = base0 + C;
  const bf16* b2p = base0 + (size_t)Win * C;
  const bf16* b3p = b2p + C;
  v8f acc0 = {}, acc1 = {}, acc2 = {}, acc3 = {};
  const bf16* wpt = wp + (size_t)mt * nKt * 512 + lane * 16;
  #pragma unroll
  for (int kt = 0; kt < nKt; ++kt) {
    v16bf A = *(const v16bf*)(wpt + kt * 512);
    v16bf B0 = load_bfrag<C>(base0, kt * 32, hi);
    acc0 = wmma_bf16(A, B0, acc0);
    v16bf B1 = load_bfrag<C>(b1p, kt * 32, hi);
    acc1 = wmma_bf16(A, B1, acc1);
    v16bf B2 = load_bfrag<C>(b2p, kt * 32, hi);
    acc2 = wmma_bf16(A, B2, acc2);
    v16bf B3 = load_bfrag<C>(b3p, kt * 32, hi);
    acc3 = wmma_bf16(A, B3, acc3);
  }
  bf16* op = out + (size_t)gp * O + mt * 16 + (hi ? 8 : 0);
  v8bf res;
  #pragma unroll
  for (int e = 0; e < 8; ++e) {
    float m = fmaxf(fmaxf(acc0[e], acc1[e]), fmaxf(acc2[e], acc3[e]));
    res[e] = (bf16)m;
  }
  *(v8bf*)op = res;
}

// ---------------------------------------------------------------------------
// plain conv1x1 GEMM, 2 N-tiles per wave (A fragment reused twice)
// act [Npix, C] -> out [Npix, O]
// ---------------------------------------------------------------------------
template<int C, int O>
__global__ void __launch_bounds__(256) gemm_conv1x1(
    const bf16* __restrict__ act, const bf16* __restrict__ wp, bf16* __restrict__ out,
    int Npix) {
  constexpr int nKt = (C + 31) / 32;
  constexpr int nMt = O / 16;
  int lane = threadIdx.x & 31;
  int hi = lane >> 4;
  int wv = (blockIdx.x * blockDim.x + threadIdx.x) >> 5;
  int nNt = Npix >> 5;            // 32 pixels per wave
  if (wv >= nMt * nNt) return;
  int mt = wv % nMt, nt = wv / nMt;
  int gp0 = nt * 32 + (lane & 15);
  const bf16* pb0 = act + (size_t)gp0 * C;
  const bf16* pb1 = pb0 + (size_t)16 * C;
  v8f acc0 = {}, acc1 = {};
  const bf16* wpt = wp + (size_t)mt * nKt * 512 + lane * 16;
  #pragma unroll
  for (int kt = 0; kt < nKt; ++kt) {
    v16bf A = *(const v16bf*)(wpt + kt * 512);
    v16bf B0 = load_bfrag<C>(pb0, kt * 32, hi);
    acc0 = wmma_bf16(A, B0, acc0);
    v16bf B1 = load_bfrag<C>(pb1, kt * 32, hi);
    acc1 = wmma_bf16(A, B1, acc1);
  }
  bf16* op0 = out + (size_t)gp0 * O + mt * 16 + (hi ? 8 : 0);
  bf16* op1 = op0 + (size_t)16 * O;
  v8bf r0, r1;
  #pragma unroll
  for (int e = 0; e < 8; ++e) { r0[e] = (bf16)acc0[e]; r1[e] = (bf16)acc1[e]; }
  *(v8bf*)op0 = r0;
  *(v8bf*)op1 = r1;
}

// ---------------------------------------------------------------------------
// hexconv = 7 shifted GEMMs on padded input, 2 N-tiles per wave.
// actP [NIMG,H+2,W+2,C] -> out [NIMG,H,W,O]
// ---------------------------------------------------------------------------
template<int C, int O, int H, int W>
__global__ void __launch_bounds__(256) gemm_hexconv(
    const bf16* __restrict__ actP, const bf16* __restrict__ wp, bf16* __restrict__ out) {
  constexpr int nKt = C / 32;
  constexpr int nMt = O / 16;
  constexpr int Npix = NIMG * H * W;
  constexpr int nNt = Npix / 32;
  constexpr int Wp = W + 2;
  // tap offsets in elements, compile-time
  constexpr int toff[7] = { (0 * Wp + 1) * C, (0 * Wp + 2) * C,
                            (1 * Wp + 0) * C, (1 * Wp + 1) * C, (1 * Wp + 2) * C,
                            (2 * Wp + 0) * C, (2 * Wp + 1) * C };
  int lane = threadIdx.x & 31;
  int hi = lane >> 4;
  int wv = (blockIdx.x * blockDim.x + threadIdx.x) >> 5;
  if (wv >= nMt * nNt) return;
  int mt = wv % nMt, nt = wv / nMt;
  int gp0 = nt * 32 + (lane & 15);
  int gp1 = gp0 + 16;
  int img0 = gp0 / (H * W), r0i = gp0 - img0 * (H * W);
  int img1 = gp1 / (H * W), r1i = gp1 - img1 * (H * W);
  int y0 = r0i / W, x0 = r0i - (r0i / W) * W;
  int y1 = r1i / W, x1 = r1i - (r1i / W) * W;
  const bf16* p0 = actP + (((size_t)img0 * (H + 2) + y0) * Wp + x0) * C;
  const bf16* p1 = actP + (((size_t)img1 * (H + 2) + y1) * Wp + x1) * C;
  v8f acc0 = {}, acc1 = {};
  const bf16* wpt = wp + (size_t)mt * nKt * 512 + lane * 16;
  #pragma unroll
  for (int tp = 0; tp < 7; ++tp) {
    const bf16* pb0 = p0 + toff[tp];
    const bf16* pb1 = p1 + toff[tp];
    const bf16* wt = wpt + (size_t)tp * nMt * nKt * 512;
    #pragma unroll
    for (int kt = 0; kt < nKt; ++kt) {
      v16bf A = *(const v16bf*)(wt + kt * 512);
      v16bf B0 = load_bfrag<C>(pb0, kt * 32, hi);
      acc0 = wmma_bf16(A, B0, acc0);
      v16bf B1 = load_bfrag<C>(pb1, kt * 32, hi);
      acc1 = wmma_bf16(A, B1, acc1);
    }
  }
  bf16* op0 = out + (size_t)gp0 * O + mt * 16 + (hi ? 8 : 0);
  bf16* op1 = out + (size_t)gp1 * O + mt * 16 + (hi ? 8 : 0);
  v8bf r0, r1;
  #pragma unroll
  for (int e = 0; e < 8; ++e) { r0[e] = (bf16)acc0[e]; r1[e] = (bf16)acc1[e]; }
  *(v8bf*)op0 = r0;
  *(v8bf*)op1 = r1;
}

// ---------------------------------------------------------------------------
// residual combine: relu(bn(c) + bn(d))
// ---------------------------------------------------------------------------
template<int C>
__global__ void combine_relu(const bf16* __restrict__ cp, const bf16* __restrict__ dp,
                             const float* __restrict__ mrc, const float* __restrict__ mrd,
                             bf16* __restrict__ out, int n) {
  int i = blockIdx.x * blockDim.x + threadIdx.x;
  if (i >= n) return;
  int c = i & (C - 1);
  float v = (float)cp[i] * mrc[c * 2] + mrc[c * 2 + 1]
          + (float)dp[i] * mrd[c * 2] + mrd[c * 2 + 1];
  out[i] = (bf16)fmaxf(v, 0.f);
}

// block2 combine + global max over charts & spatial -> feat[b*C+c]
template<int C, int HW>
__global__ void combine_max_kernel(const bf16* __restrict__ cp, const bf16* __restrict__ dp,
                                   const float* __restrict__ mrc, const float* __restrict__ mrd,
                                   float* __restrict__ feat) {
  int b = blockIdx.x;
  int c = threadIdx.x;
  float sc = mrc[c * 2], sh = mrc[c * 2 + 1];
  float sd = mrd[c * 2], shd = mrd[c * 2 + 1];
  float mx = 0.f;  // post-ReLU values are >= 0
  for (int f = 0; f < 5; ++f) {
    size_t base = ((size_t)(f * NB + b) * HW) * C + c;
    #pragma unroll 4
    for (int p = 0; p < HW; ++p) {
      float v = (float)cp[base] * sc + sh + (float)dp[base] * sd + shd;
      v = fmaxf(v, 0.f);
      mx = fmaxf(mx, v);
      base += C;
    }
  }
  feat[b * C + c] = mx;
}

__global__ void fc_kernel(const float* __restrict__ feat, const float* __restrict__ w,
                          const float* __restrict__ bias, float* __restrict__ out) {
  int i = blockIdx.x * blockDim.x + threadIdx.x;
  if (i >= NB * 10) return;
  int o = i % 10, b = i / 10;
  float s = bias[o];
  const float* fb = feat + (size_t)b * 256;
  const float* wb = w + (size_t)o * 256;
  for (int c = 0; c < 256; ++c) s += fb[c] * wb[c];
  out[i] = s;
}

// ---------------------------------------------------------------------------
extern "C" void kernel_launch(void* const* d_in, const int* in_sizes, int n_in,
                              void* d_out, int out_size, void* d_ws, size_t ws_size,
                              hipStream_t stream) {
  (void)in_sizes; (void)n_in; (void)out_size; (void)ws_size;
  const float* x       = (const float*)d_in[0];
  const float* w_conv1 = (const float*)d_in[1];
  const float* g1   = (const float*)d_in[2];
  const float* bb1  = (const float*)d_in[3];
  const float* b1_c1w  = (const float*)d_in[4];
  const float* b1_hexw = (const float*)d_in[5];
  const float* b1_c3w  = (const float*)d_in[6];
  const float* b1_dsw  = (const float*)d_in[7];
  const float* b1_ga = (const float*)d_in[8];
  const float* b1_ba = (const float*)d_in[9];
  const float* b1_gb = (const float*)d_in[10];
  const float* b1_bb = (const float*)d_in[11];
  const float* b1_gc = (const float*)d_in[12];
  const float* b1_bc = (const float*)d_in[13];
  const float* b1_gd = (const float*)d_in[14];
  const float* b1_bd = (const float*)d_in[15];
  const float* b2_c1w  = (const float*)d_in[16];
  const float* b2_hexw = (const float*)d_in[17];
  const float* b2_c3w  = (const float*)d_in[18];
  const float* b2_dsw  = (const float*)d_in[19];
  const float* b2_ga = (const float*)d_in[20];
  const float* b2_ba = (const float*)d_in[21];
  const float* b2_gb = (const float*)d_in[22];
  const float* b2_bb = (const float*)d_in[23];
  const float* b2_gc = (const float*)d_in[24];
  const float* b2_bc = (const float*)d_in[25];
  const float* b2_gd = (const float*)d_in[26];
  const float* b2_bd = (const float*)d_in[27];
  const float* fc_w = (const float*)d_in[28];
  const float* fc_b = (const float*)d_in[29];
  float* outp = (float*)d_out;

  // workspace arena (all offsets 256B aligned) -- ~139 MB total
  char* cur = (char*)d_ws;
  auto alloc = [&](size_t bytes) -> char* {
    char* p = cur; cur += (bytes + 255) & ~(size_t)255; return p;
  };
  bf16* pw_b1c1  = (bf16*)alloc(2048 * 2);
  bf16* pw_b1ds  = (bf16*)alloc(2048 * 2);
  bf16* pw_b1c3  = (bf16*)alloc(4096 * 2);
  bf16* pw_b1hex = (bf16*)alloc(28672 * 2);
  bf16* pw_b2c1  = (bf16*)alloc(16384 * 2);
  bf16* pw_b2ds  = (bf16*)alloc(16384 * 2);
  bf16* pw_b2c3  = (bf16*)alloc(65536 * 2);
  bf16* pw_b2hex = (bf16*)alloc(458752 * 2);
  float* st_part = (float*)alloc((size_t)NSLICE * 256 * 2 * 4);
  float* mr0     = (float*)alloc(256 * 2 * 4);
  float* mr1     = (float*)alloc(256 * 2 * 4);
  const int ELB = 10485760;  // every big tensor here is 10,485,760 elems
  bf16* bufA = (bf16*)alloc((size_t)ELB * 2);  // t0 -> h_pre -> a2_pre -> c2_pre
  bf16* bufB = (bf16*)alloc((size_t)ELB * 2);  // t0bn -> h -> h2_pre
  bf16* bufC = (bf16*)alloc((size_t)ELB * 2);  // a_pre -> c_pre -> h2
  bf16* bufD = (bf16*)alloc((size_t)ELB * 2);  // d_pre -> d2_pre
  bf16* bufF = (bf16*)alloc((size_t)ELB * 2);  // out1
  bf16* bufE = (bf16*)alloc((size_t)19660800 * 2);  // padded (max of block1/2)
  float* feat = (float*)alloc((size_t)NB * 256 * 4);

  dim3 B256(256);
  auto g1dim = [](int n) { return dim3((unsigned)((n + 255) / 256)); };

  // ---- pack weights into WMMA fragment layout ------------------------------
  pack_gemm_w<<<g1dim(2048),   B256, 0, stream>>>(b1_c1w, pw_b1c1, 64, 16);
  pack_gemm_w<<<g1dim(2048),   B256, 0, stream>>>(b1_dsw, pw_b1ds, 64, 16);
  pack_gemm_w<<<g1dim(4096),   B256, 0, stream>>>(b1_c3w, pw_b1c3, 64, 64);
  pack_hex_w <<<g1dim(28672),  B256, 0, stream>>>(b1_hexw, pw_b1hex, 64, 64);
  pack_gemm_w<<<g1dim(16384),  B256, 0, stream>>>(b2_c1w, pw_b2c1, 256, 64);
  pack_gemm_w<<<g1dim(16384),  B256, 0, stream>>>(b2_dsw, pw_b2ds, 256, 64);
  pack_gemm_w<<<g1dim(65536),  B256, 0, stream>>>(b2_c3w, pw_b2c3, 256, 256);
  pack_hex_w <<<g1dim(458752), B256, 0, stream>>>(b2_hexw, pw_b2hex, 256, 256);

  // ---- stem: relu(hexconv) -> BN -------------------------------------------
  stem_kernel<<<g1dim(ELB), B256, 0, stream>>>(x, w_conv1, bufA);
  stats_partial<<<dim3(16 * NSLICE), B256, 0, stream>>>(bufA, st_part, NIMG * 512, 16);
  stats_final<<<dim3(1), B256, 0, stream>>>(st_part, g1, bb1, mr0, 16, 1.0f / (NIMG * 512));
  normalize_plain<16><<<g1dim(ELB), B256, 0, stream>>>(bufA, mr0, bufB, ELB, 0);

  // ---- block1: conv1x1+pool (a path & ds path) -----------------------------
  {
    int waves = 4 * (NIMG * 8 * 16 / 16);   // 40960
    gemm_conv_pool<16, 16, 32, 64><<<g1dim(waves * 32), B256, 0, stream>>>(bufB, pw_b1c1, bufC);
    gemm_conv_pool<16, 16, 32, 64><<<g1dim(waves * 32), B256, 0, stream>>>(bufB, pw_b1ds, bufD);
  }
  // bn_a + relu + pad for hexconv
  stats_partial<<<dim3(64 * NSLICE), B256, 0, stream>>>(bufC, st_part, NIMG * 128, 64);
  stats_final<<<dim3(1), B256, 0, stream>>>(st_part, b1_ga, b1_ba, mr0, 64, 1.0f / (NIMG * 128));
  clear_bf16<<<g1dim(14745600), B256, 0, stream>>>(bufE, 14745600);
  normalize_pad<64, 8, 16><<<g1dim(ELB), B256, 0, stream>>>(bufC, mr0, bufE);
  halo_fill<64, 8, 16><<<g1dim(NIMG * 8 * 64 * 2), B256, 0, stream>>>(bufE);
  // hexconv
  {
    int waves = 4 * (NIMG * 128 / 32);      // 20480 (2 N-tiles per wave)
    gemm_hexconv<64, 64, 8, 16><<<g1dim(waves * 32), B256, 0, stream>>>(bufE, pw_b1hex, bufA);
  }
  stats_partial<<<dim3(64 * NSLICE), B256, 0, stream>>>(bufA, st_part, NIMG * 128, 64);
  stats_final<<<dim3(1), B256, 0, stream>>>(st_part, b1_gb, b1_bb, mr0, 64, 1.0f / (NIMG * 128));
  normalize_plain<64><<<g1dim(ELB), B256, 0, stream>>>(bufA, mr0, bufB, ELB, 1);
  // conv1x1 (c path)
  {
    int waves = 4 * (NIMG * 128 / 32);
    gemm_conv1x1<64, 64><<<g1dim(waves * 32), B256, 0, stream>>>(bufB, pw_b1c3, bufC, NIMG * 128);
  }
  stats_partial<<<dim3(64 * NSLICE), B256, 0, stream>>>(bufC, st_part, NIMG * 128, 64);
  stats_final<<<dim3(1), B256, 0, stream>>>(st_part, b1_gc, b1_bc, mr0, 64, 1.0f / (NIMG * 128));
  stats_partial<<<dim3(64 * NSLICE), B256, 0, stream>>>(bufD, st_part, NIMG * 128, 64);
  stats_final<<<dim3(1), B256, 0, stream>>>(st_part, b1_gd, b1_bd, mr1, 64, 1.0f / (NIMG * 128));
  combine_relu<64><<<g1dim(ELB), B256, 0, stream>>>(bufC, bufD, mr0, mr1, bufF, ELB);

  // ---- block2 --------------------------------------------------------------
  {
    int waves = 16 * (NIMG * 4 * 8 / 16);   // 40960
    gemm_conv_pool<64, 8, 16, 256><<<g1dim(waves * 32), B256, 0, stream>>>(bufF, pw_b2c1, bufA);
    gemm_conv_pool<64, 8, 16, 256><<<g1dim(waves * 32), B256, 0, stream>>>(bufF, pw_b2ds, bufD);
  }
  stats_partial<<<dim3(256 * NSLICE), B256, 0, stream>>>(bufA, st_part, NIMG * 32, 256);
  stats_final<<<dim3(1), B256, 0, stream>>>(st_part, b2_ga, b2_ba, mr0, 256, 1.0f / (NIMG * 32));
  clear_bf16<<<g1dim(19660800), B256, 0, stream>>>(bufE, 19660800);
  normalize_pad<256, 4, 8><<<g1dim(ELB), B256, 0, stream>>>(bufA, mr0, bufE);
  halo_fill<256, 4, 8><<<g1dim(NIMG * 4 * 256 * 2), B256, 0, stream>>>(bufE);
  {
    int waves = 16 * (NIMG * 32 / 32);      // 20480
    gemm_hexconv<256, 256, 4, 8><<<g1dim(waves * 32), B256, 0, stream>>>(bufE, pw_b2hex, bufB);
  }
  stats_partial<<<dim3(256 * NSLICE), B256, 0, stream>>>(bufB, st_part, NIMG * 32, 256);
  stats_final<<<dim3(1), B256, 0, stream>>>(st_part, b2_gb, b2_bb, mr0, 256, 1.0f / (NIMG * 32));
  normalize_plain<256><<<g1dim(ELB), B256, 0, stream>>>(bufB, mr0, bufC, ELB, 1);
  {
    int waves = 16 * (NIMG * 32 / 32);
    gemm_conv1x1<256, 256><<<g1dim(waves * 32), B256, 0, stream>>>(bufC, pw_b2c3, bufA, NIMG * 32);
  }
  stats_partial<<<dim3(256 * NSLICE), B256, 0, stream>>>(bufA, st_part, NIMG * 32, 256);
  stats_final<<<dim3(1), B256, 0, stream>>>(st_part, b2_gc, b2_bc, mr0, 256, 1.0f / (NIMG * 32));
  stats_partial<<<dim3(256 * NSLICE), B256, 0, stream>>>(bufD, st_part, NIMG * 32, 256);
  stats_final<<<dim3(1), B256, 0, stream>>>(st_part, b2_gd, b2_bd, mr1, 256, 1.0f / (NIMG * 32));

  // combine + global max over (chart, H, W), then FC
  combine_max_kernel<256, 32><<<dim3(NB), B256, 0, stream>>>(bufA, bufD, mr0, mr1, feat);
  fc_kernel<<<g1dim(NB * 10), B256, 0, stream>>>(feat, fc_w, fc_b, outp);
}
